// RNNModel_5102421147662
// MI455X (gfx1250) — 16-bit and 8-bit matrix operand order repaired and checked by reading, compiled, not run on hardware
//
#include <hip/hip_runtime.h>
#include <stdint.h>


#define V_SZ 50255
#define E_SZ 128
#define H_SZ 256
#define B_SZ 16
#define T_SZ 256
#define G4H  1024

typedef __attribute__((ext_vector_type(16))) __bf16        v16bf;
typedef __attribute__((ext_vector_type(8)))  float         v8f;
typedef __attribute__((ext_vector_type(8)))  int           v8i;
typedef __attribute__((ext_vector_type(4)))  unsigned int  u32x4;
typedef __attribute__((ext_vector_type(2)))  unsigned int  u32x2;

union Frag16 { v16bf v; u32x4 u[2]; };
union Frag8  { v8i   v; u32x4 u[2]; u32x2 c[4]; };

__device__ __forceinline__ unsigned short f32_to_bf16(float f) {
  unsigned int b = __float_as_uint(f);
  unsigned int r = b + 0x7FFFu + ((b >> 16) & 1u);
  return (unsigned short)(r >> 16);
}

__device__ __forceinline__ unsigned char f32_to_fp8(float f) {
  unsigned int u = __float_as_uint(f);
  unsigned int s = (u >> 24) & 0x80u;
  int exp = (int)((u >> 23) & 0xFFu) - 127;
  unsigned int man = u & 0x7FFFFFu;
  if (exp > 8 || (exp == 8 && man > 0x600000u)) return (unsigned char)(s | 0x7Eu);
  if (exp < -9) return (unsigned char)s;
  if (exp >= -6) {
    unsigned int m3  = man >> 20;
    unsigned int rem = man & 0xFFFFFu;
    if (rem > 0x80000u || (rem == 0x80000u && (m3 & 1u))) m3++;
    unsigned int e = (unsigned int)(exp + 7);
    if (m3 == 8u) { m3 = 0u; e++; if (e >= 16u) return (unsigned char)(s | 0x7Eu); }
    return (unsigned char)(s | (e << 3) | m3);
  }
  unsigned int full  = man | 0x800000u;
  int shift = 20 + (-6 - exp);
  unsigned int m3   = full >> shift;
  unsigned int rem  = full & ((1u << shift) - 1u);
  unsigned int half = 1u << (shift - 1);
  if (rem > half || (rem == half && (m3 & 1u))) m3++;
  if (m3 >= 8u) return (unsigned char)(s | (1u << 3));
  return (unsigned char)(s | m3);
}

#define LOG2E_F 1.4426950408889634f

__device__ __forceinline__ float sigmoidf_fast(float x) {
  float e = __builtin_amdgcn_exp2f(-LOG2E_F * x);
  return __builtin_amdgcn_rcpf(1.0f + e);
}

__device__ __forceinline__ float tanhf_fast(float x) {
#if __has_builtin(__builtin_amdgcn_tanhf)
  return __builtin_amdgcn_tanhf(x);
#else
  float e = __builtin_amdgcn_exp2f(2.0f * LOG2E_F * x);
  return 1.0f - 2.0f * __builtin_amdgcn_rcpf(e + 1.0f);
#endif
}

__global__ void convert_bf16_kernel(const float* __restrict__ src,
                                    unsigned short* __restrict__ dst, int n) {
  int i = blockIdx.x * blockDim.x + threadIdx.x;
  int stride = gridDim.x * blockDim.x;
  for (; i < n; i += stride) dst[i] = f32_to_bf16(src[i]);
}

__global__ void convert_fp8_kernel(const float* __restrict__ src,
                                   unsigned char* __restrict__ dst, int n) {
  int i = blockIdx.x * blockDim.x + threadIdx.x;
  int stride = gridDim.x * blockDim.x;
  for (; i < n; i += stride) dst[i] = f32_to_fp8(src[i]);
}

template <int K, bool GATHER>
__global__ void __launch_bounds__(256)
wmma_gemm_kernel(const unsigned short* __restrict__ A,
                 const int* __restrict__ idx,
                 const unsigned short* __restrict__ W,
                 const float* __restrict__ bias0,
                 const float* __restrict__ bias1,
                 float* __restrict__ out,
                 int N, int ldN) {
  __shared__ __align__(16) unsigned short sB[128 * K];

  const int tid = threadIdx.x;
  const int n0  = blockIdx.x * 128;

  for (int i = tid; i < 16 * K; i += 256) {
    const int col = i / (K / 8);
    const int kk  = (i % (K / 8)) * 8;
    unsigned short* lp = sB + col * K + kk;
    if (n0 + col < N) {
      const unsigned short* gp = W + (size_t)(n0 + col) * K + kk;
      unsigned lds_addr = (unsigned)(unsigned long long)(void*)lp;
      unsigned long long gaddr = (unsigned long long)(void*)gp;
      asm volatile("global_load_async_to_lds_b128 %0, %1, off offset:0"
                   :: "v"(lds_addr), "v"(gaddr) : "memory");
    } else {
      u32x4 z = {0u, 0u, 0u, 0u};
      *(u32x4*)lp = z;
    }
  }
  asm volatile("s_wait_asynccnt 0x0" ::: "memory");
  __syncthreads();

  const int wave  = tid >> 5;
  const int lane  = tid & 31;
  const int lhalf = lane >> 4;
  const int l16   = lane & 15;
  const int m0    = blockIdx.y * 128 + wave * 16;

  v8f acc[8] = {};

  #pragma unroll
  for (int ks = 0; ks < K / 32; ++ks) {
    const unsigned short* arow =
        GATHER ? (A + (size_t)idx[m0 + l16] * K)
               : (A + (size_t)(m0 + l16) * K);
    const int kbase = ks * 32 + 8 * lhalf;
    Frag16 af;
    af.u[0] = *(const u32x4*)(arow + kbase);
    af.u[1] = *(const u32x4*)(arow + kbase + 16);

    #pragma unroll
    for (int nt = 0; nt < 8; ++nt) {
      const unsigned short* brow =
          sB + (nt * 16 + l16) * K + ks * 32 + 8 * lhalf;
      Frag16 bf;
      bf.u[0] = *(const u32x4*)(brow);
      bf.u[1] = *(const u32x4*)(brow + 16);
      acc[nt] = __builtin_amdgcn_wmma_f32_16x16x32_bf16(
          false, af.v, false, bf.v, (short)0, acc[nt], false, false);
    }
  }

  #pragma unroll
  for (int nt = 0; nt < 8; ++nt) {
    const int n = n0 + nt * 16 + l16;
    if (n < N) {
      float b = 0.0f;
      if (bias0) b += bias0[n];
      if (bias1) b += bias1[n];
      #pragma unroll
      for (int j = 0; j < 8; ++j) {
        const int row = m0 + j + 8 * lhalf;
        out[(size_t)row * ldN + n] = acc[nt][j] + b;
      }
    }
  }
}

__global__ void __launch_bounds__(512)
lstm_kernel(const float* __restrict__ xg,
            const unsigned char* __restrict__ Whh8,
            unsigned short* __restrict__ hs) {
  __shared__ __align__(16) unsigned char sH[B_SZ * H_SZ];

  const int tid   = threadIdx.x;
  const int wave  = tid >> 5;
  const int lane  = tid & 31;
  const int lhalf = lane >> 4;
  const int l16   = lane & 15;
  const int ncol  = wave * 16 + l16;

  Frag8 bfr[16];
  #pragma unroll
  for (int g = 0; g < 4; ++g) {
    #pragma unroll
    for (int ks = 0; ks < 4; ++ks) {
      const unsigned char* brow =
          Whh8 + (size_t)(g * H_SZ + wave * 16 + l16) * H_SZ + ks * 64;
      bfr[g * 4 + ks].u[0] = *(const u32x4*)(brow + 16 * lhalf);
      bfr[g * 4 + ks].u[1] = *(const u32x4*)(brow + 32 + 16 * lhalf);
    }
  }

  for (int i = tid; i < B_SZ * H_SZ; i += 512) sH[i] = 0;
  __syncthreads();

  float cr[8];
  #pragma unroll
  for (int j = 0; j < 8; ++j) cr[j] = 0.0f;

  for (int t = 0; t < T_SZ; ++t) {
    v8f acc[4] = {};
    #pragma unroll
    for (int ks = 0; ks < 4; ++ks) {
      Frag8 af;
      const unsigned char* hrow = sH + l16 * H_SZ + ks * 64;
      #pragma unroll
      for (int c = 0; c < 4; ++c)
        af.c[c] = *(const u32x2*)(hrow + 32 * (c >> 1) + 8 * (c & 1) + 16 * lhalf);
      #pragma unroll
      for (int g = 0; g < 4; ++g)
        acc[g] = __builtin_amdgcn_wmma_f32_16x16x64_fp8_fp8(
            af.v, bfr[g * 4 + ks].v, (short)0, acc[g], false, false);
    }
    __syncthreads();

    #pragma unroll
    for (int j = 0; j < 8; ++j) {
      const int m = j + 8 * lhalf;
      const size_t xrow = ((size_t)m * T_SZ + t) * G4H;
      float gi = sigmoidf_fast(acc[0][j] + xg[xrow + 0 * H_SZ + ncol]);
      float gf = sigmoidf_fast(acc[1][j] + xg[xrow + 1 * H_SZ + ncol]);
      float gg = tanhf_fast   (acc[2][j] + xg[xrow + 2 * H_SZ + ncol]);
      float go = sigmoidf_fast(acc[3][j] + xg[xrow + 3 * H_SZ + ncol]);
      cr[j] = gf * cr[j] + gi * gg;
      const float h = go * tanhf_fast(cr[j]);
      sH[m * H_SZ + ncol] = f32_to_fp8(h);
      hs[((size_t)m * T_SZ + t) * H_SZ + ncol] = f32_to_bf16(h);
    }
    __syncthreads();
  }
}

extern "C" void kernel_launch(void* const* d_in, const int* in_sizes, int n_in,
                              void* d_out, int out_size, void* d_ws, size_t ws_size,
                              hipStream_t stream) {
  const int*   x    = (const int*)  d_in[0];
  const float* emb  = (const float*)d_in[1];
  const float* W_ih = (const float*)d_in[2];
  const float* W_hh = (const float*)d_in[3];
  const float* b_ih = (const float*)d_in[4];
  const float* b_hh = (const float*)d_in[5];
  const float* W_fc = (const float*)d_in[6];
  const float* b_fc = (const float*)d_in[7];
  float* out = (float*)d_out;

  char* ws = (char*)d_ws;
  size_t off = 0;
  auto carve = [&](size_t bytes) -> void* {
    void* p = ws + off;
    off += (bytes + 255) & ~(size_t)255;
    return p;
  };
  unsigned short* emb_bf = (unsigned short*)carve((size_t)V_SZ * E_SZ * 2);
  unsigned short* wih_bf = (unsigned short*)carve((size_t)G4H  * E_SZ * 2);
  unsigned char*  whh_f8 = (unsigned char*) carve((size_t)G4H  * H_SZ);
  unsigned short* wfc_bf = (unsigned short*)carve((size_t)V_SZ * H_SZ * 2);
  float*          xg     = (float*)         carve((size_t)B_SZ * T_SZ * G4H * 4);
  unsigned short* hs     = (unsigned short*)carve((size_t)B_SZ * T_SZ * H_SZ * 2);

  convert_bf16_kernel<<<dim3(1024), dim3(256), 0, stream>>>(emb,  emb_bf, V_SZ * E_SZ);
  convert_bf16_kernel<<<dim3(256),  dim3(256), 0, stream>>>(W_ih, wih_bf, G4H * E_SZ);
  convert_fp8_kernel <<<dim3(256),  dim3(256), 0, stream>>>(W_hh, whh_f8, G4H * H_SZ);
  convert_bf16_kernel<<<dim3(1024), dim3(256), 0, stream>>>(W_fc, wfc_bf, V_SZ * H_SZ);

  wmma_gemm_kernel<E_SZ, true>
      <<<dim3(G4H / 128, (B_SZ * T_SZ) / 128), dim3(256), 0, stream>>>(
          emb_bf, x, wih_bf, b_ih, b_hh, xg, G4H, G4H);

  lstm_kernel<<<dim3(1), dim3(512), 0, stream>>>(xg, whh_f8, hs);

  wmma_gemm_kernel<H_SZ, false>
      <<<dim3((V_SZ + 127) / 128, (B_SZ * T_SZ) / 128), dim3(256), 0, stream>>>(
          hs, nullptr, wfc_bf, b_fc, nullptr, out, V_SZ, V_SZ);
}
